// FeedForward_40664750358804
// MI455X (gfx1250) — matrix operand order repaired and checked by reading, not compiled, not run on hardware
//
#include <hip/hip_runtime.h>
#include <math.h>

// ---------------------------------------------------------------------------
// Fused gated conv-FFN for MI455X (gfx1250, wave32, WMMA).
//   y = W_out @ ( gelu(dw3x3(W_in@x)[0:340]) * dw3x3(W_in@x)[340:680] )
// One kernel, one pass over HBM: x in (67MB) + y out (67MB); all 680-channel
// intermediates live in LDS per 16x16 spatial tile. Pointwise convs run as
// bf16 WMMA GEMMs (16x16x32), depthwise+GELU on the VALU.
// ---------------------------------------------------------------------------

typedef __bf16 bf16_t;
typedef __attribute__((ext_vector_type(16))) __bf16 v16bf;
typedef __attribute__((ext_vector_type(8)))  float  v8f;
typedef __attribute__((ext_vector_type(4)))  float  v4f;

namespace {

constexpr int DIM    = 128;
constexpr int HID    = 340;
constexpr int HIDP   = 352;          // hidden padded to multiple of 32
constexpr int CH     = 32;           // hidden channels per chunk
constexpr int NCHUNK = HIDP / CH;    // 11
constexpr int NB     = 2;
constexpr int IMG_H  = 256, IMG_W = 256;
constexpr int TH = 16, TW = 16;      // output tile
constexpr int NPIX   = TH * TW;      // 256
constexpr int HALO_W = TW + 2;       // 18
constexpr int NHALO  = HALO_W * (TH + 2);  // 324
constexpr int NHALO_P = 336;         // halo pixels padded to /16 (21 N-tiles)

// LDS row strides in bf16 elements (+8 pad: 16B-aligned, bank-staggered)
constexpr int XK   = DIM + 8;        // 136  Xs: [pixel][k]      336 rows
constexpr int WA_K = DIM + 8;        // 136  Wa: [row][k]         64 rows
constexpr int HROW = NHALO_P + 8;    // 344  Hs: [ch][pixel]      64 rows
constexpr int GK   = CH + 8;         // 40   Gs: [pixel][k]      256 rows
constexpr int WO_K = CH + 8;         // 40   Wo: [row][k]        128 rows

constexpr int SM_X  = 0;
constexpr int SM_WA = SM_X  + NHALO_P * XK;
constexpr int SM_H  = SM_WA + 2 * CH * WA_K;
constexpr int SM_G  = SM_H  + 2 * CH * HROW;
constexpr int SM_WO = SM_G  + NPIX * GK;
constexpr int SM_TOT = SM_WO + DIM * WO_K;   // 91776 bf16 = 183552 B < 320 KB

union Frag { v16bf v; v4f q[2]; };

// A-matrix 16x32 bf16 fragment (ISA 7.12.2): lane m = l&15,
// lanes 0-15: elems 0-7 -> K 0-7,  8-15 -> K 16-23
// lanes16-31: elems 0-7 -> K 8-15, 8-15 -> K 24-31
__device__ inline v16bf ld_frag_a(const bf16_t* base, int rs, int kbase, int lane) {
    const int m  = lane & 15;
    const int ko = (lane >> 4) << 3;             // 0 or 8
    const bf16_t* p = base + m * rs + kbase + ko;
    Frag f;
    f.q[0] = *(const v4f*)(p);                   // K = kbase+ko   .. +7
    f.q[1] = *(const v4f*)(p + 16);              // K = kbase+ko+16.. +23
    return f.v;
}

// B-matrix 32x16 bf16 fragment: lane n = l&15,
// lanes 0-15: elems 0-7 -> K 0-7, 8-15 -> K 16-23; lanes 16-31: K 8-15, K 24-31
__device__ inline v16bf ld_frag_b(const bf16_t* base, int rs, int kbase, int lane) {
    const int n  = lane & 15;
    const int ko = (lane >> 4) << 3;             // 0 or 8
    const bf16_t* p = base + n * rs + kbase + ko;
    Frag f;
    f.q[0] = *(const v4f*)(p);
    f.q[1] = *(const v4f*)(p + 16);
    return f.v;
}

__device__ inline v8f v8f_zero() {
    v8f z;
#pragma unroll
    for (int i = 0; i < 8; ++i) z[i] = 0.0f;
    return z;
}

} // namespace

__global__ __launch_bounds__(256, 1)
void ffn_fused_kernel(const float* __restrict__ x,
                      const float* __restrict__ w_in,
                      const float* __restrict__ w_dw,
                      const float* __restrict__ w_out,
                      float* __restrict__ y)
{
    extern __shared__ bf16_t smem[];
    bf16_t* Xs = smem + SM_X;    // [336][136]  x halo, bf16, [pixel][k]
    bf16_t* Wa = smem + SM_WA;   // [ 64][136]  W_in chunk rows (h1|h2)
    bf16_t* Hs = smem + SM_H;    // [ 64][344]  h chunk, [ch][halo pixel]
    bf16_t* Gs = smem + SM_G;    // [256][ 40]  gated activations, [pixel][k]
    bf16_t* Wo = smem + SM_WO;   // [128][ 40]  W_out chunk

    const int tid  = threadIdx.x;
    const int lane = tid & 31;
    const int wave = tid >> 5;                         // 0..7
    const int tiles_x = IMG_W / TW;                    // 16
    const int tiles_y = IMG_H / TH;                    // 16
    const int b   = blockIdx.x / (tiles_x * tiles_y);
    const int tr  = blockIdx.x % (tiles_x * tiles_y);
    const int ty0 = (tr / tiles_x) * TH;
    const int tx0 = (tr % tiles_x) * TW;

    // ---- Stage 0: x halo (128 ch x 18x18) -> LDS bf16, zero-padded to 336 px
    for (int idx = tid; idx < DIM * NHALO_P; idx += 256) {
        const int c = idx / NHALO_P;
        const int p = idx % NHALO_P;
        float v = 0.0f;
        if (p < NHALO) {
            const int gy = ty0 + (p / HALO_W) - 1;
            const int gx = tx0 + (p % HALO_W) - 1;
            if ((unsigned)gy < (unsigned)IMG_H && (unsigned)gx < (unsigned)IMG_W)
                v = x[((b * DIM + c) * IMG_H + gy) * IMG_W + gx];
        }
        Xs[p * XK + c] = (bf16_t)v;
    }
    __syncthreads();

    // Y accumulators: this wave owns pixel rows {2*wave, 2*wave+1} x 8 ch-tiles
    v8f acc[16];
#pragma unroll
    for (int i = 0; i < 16; ++i) acc[i] = v8f_zero();

    for (int chunk = 0; chunk < NCHUNK; ++chunk) {
        const int c0 = chunk * CH;                     // padded hidden base

        // hint: pull next chunk's weights toward L2 (global_prefetch_b8)
        if (chunk + 1 < NCHUNK && tid < 64) {
            const int nc0 = (chunk + 1) * CH;
            __builtin_prefetch(&w_in[(nc0 + tid) * DIM], 0, 1);
            __builtin_prefetch(&w_out[(tid * 2) * HID + nc0], 0, 1);
        }

        // ---- Stage 1: stage W_in rows (32 of half1, 32 of half2) as bf16
        for (int idx = tid; idx < 2 * CH * DIM; idx += 256) {
            const int r = idx / DIM;
            const int k = idx % DIM;
            const int hc = (r < CH) ? (c0 + r) : (c0 + r - CH);
            float v = 0.0f;
            if (hc < HID) {
                const int row = (r < CH) ? hc : (HID + hc);
                v = w_in[row * DIM + k];
            }
            Wa[r * WA_K + k] = (bf16_t)v;
        }
        __syncthreads();

        // ---- Stage 2: GEMM1  h[64 x 336] = Wa[64 x 128] @ X[128 x 336]
        for (int nt = wave; nt < NHALO_P / 16; nt += 8) {
#pragma unroll
            for (int mt = 0; mt < 4; ++mt) {           // 4 tiles of 16 channels
                v8f c = v8f_zero();
#pragma unroll
                for (int kb = 0; kb < DIM; kb += 32) {
                    v16bf af = ld_frag_a(Wa + (mt * 16) * WA_K, WA_K, kb, lane);
                    v16bf bf = ld_frag_b(Xs + (nt * 16) * XK,  XK,  kb, lane);
                    c = __builtin_amdgcn_wmma_f32_16x16x32_bf16(
                            false, af, false, bf, (short)0, c, false, false);
                }
                const int n  = lane & 15;
                const int mb = (lane >> 4) << 3;
#pragma unroll
                for (int r = 0; r < 8; ++r)
                    Hs[(mt * 16 + mb + r) * HROW + nt * 16 + n] = (bf16_t)c[r];
            }
        }
        __syncthreads();

        // ---- Stage 3: depthwise 3x3 + exact-erf GELU gate -> Gs[pixel][k]
        {
            const int ch    = tid >> 3;                // 0..31 local hidden ch
            const bool ok   = (c0 + ch) < HID;
            const int hc1   = c0 + ch;
            const int hc2   = HID + c0 + ch;
            float wd1[9], wd2[9];
#pragma unroll
            for (int t = 0; t < 9; ++t) {
                wd1[t] = ok ? w_dw[hc1 * 9 + t] : 0.0f;
                wd2[t] = ok ? w_dw[hc2 * 9 + t] : 0.0f;
            }
            for (int p = (tid & 7); p < NPIX; p += 8) {
                const int py = p >> 4, px = p & 15;
                float d1 = 0.0f, d2 = 0.0f;
#pragma unroll
                for (int dy = 0; dy < 3; ++dy)
#pragma unroll
                    for (int dx = 0; dx < 3; ++dx) {
                        const int hp = (py + dy) * HALO_W + (px + dx);
                        d1 += wd1[dy * 3 + dx] * (float)Hs[ch * HROW + hp];
                        d2 += wd2[dy * 3 + dx] * (float)Hs[(CH + ch) * HROW + hp];
                    }
                const float g = 0.5f * d1 * (1.0f + erff(d1 * 0.70710678f)) * d2;
                Gs[p * GK + ch] = (bf16_t)g;
            }
        }
        __syncthreads();

        // ---- Stage 4: stage W_out chunk [128][32] as bf16 (zero-padded K)
        for (int idx = tid; idx < DIM * CH; idx += 256) {
            const int r = idx / CH;
            const int k = idx % CH;
            const float v = (c0 + k < HID) ? w_out[r * HID + c0 + k] : 0.0f;
            Wo[r * WO_K + k] = (bf16_t)v;
        }
        __syncthreads();

        // ---- Stage 5: GEMM2 accumulate  Y[128 x 256] += Wo[128 x 32] @ G
#pragma unroll
        for (int i = 0; i < 2; ++i) {
            const int nt = wave * 2 + i;
            v16bf bf = ld_frag_b(Gs + (nt * 16) * GK, GK, 0, lane);
#pragma unroll
            for (int mt = 0; mt < 8; ++mt) {
                v16bf af = ld_frag_a(Wo + (mt * 16) * WO_K, WO_K, 0, lane);
                acc[i * 8 + mt] = __builtin_amdgcn_wmma_f32_16x16x32_bf16(
                        false, af, false, bf, (short)0, acc[i * 8 + mt],
                        false, false);
            }
        }
        __syncthreads();   // before next chunk overwrites Wa/Hs/Gs/Wo
    }

    // ---- Epilogue: write Y tile (coalesced 16-lane fp32 rows)
    {
        const int n  = lane & 15;
        const int mb = (lane >> 4) << 3;
#pragma unroll
        for (int i = 0; i < 2; ++i) {
            const int p  = (wave * 2 + i) * 16 + n;    // p = ty*16 + tx
            const int gy = ty0 + (p >> 4);
            const int gx = tx0 + (p & 15);
#pragma unroll
            for (int mt = 0; mt < 8; ++mt)
#pragma unroll
                for (int r = 0; r < 8; ++r) {
                    const int chn = mt * 16 + mb + r;
                    y[((b * DIM + chn) * IMG_H + gy) * IMG_W + gx] =
                        acc[i * 8 + mt][r];
                }
        }
    }
}

extern "C" void kernel_launch(void* const* d_in, const int* in_sizes, int n_in,
                              void* d_out, int out_size, void* d_ws, size_t ws_size,
                              hipStream_t stream) {
    (void)in_sizes; (void)n_in; (void)out_size; (void)d_ws; (void)ws_size;
    const float* x     = (const float*)d_in[0];
    const float* w_in  = (const float*)d_in[1];
    const float* w_dw  = (const float*)d_in[2];
    const float* w_out = (const float*)d_in[3];
    float*       yout  = (float*)d_out;

    const int grid = NB * (IMG_H / TH) * (IMG_W / TW);   // 512 workgroups
    const size_t shmem = (size_t)SM_TOT * sizeof(bf16_t); // ~183.5 KB dynamic LDS
    ffn_fused_kernel<<<grid, 256, shmem, stream>>>(x, w_in, w_dw, w_out, yout);
}
